// GCN_8_plus_6408091205937
// MI455X (gfx1250) — matrix operand order repaired and checked by reading, not compiled, not run on hardware
//
#include <hip/hip_runtime.h>
#include <hip/hip_bf16.h>

// ---------------------------------------------------------------------------
// GAT (3-layer, edge-featured) forward for MI455X (gfx1250).
// Dense projections via v_wmma_f32_16x16x32_f16 (wave32 WMMA), fragment-order
// pre-swizzled B (all fragment loads = constant-offset b128), b128 A loads,
// compile-time M/N/K addressing. Edge attention via fused per-edge kernels +
// L2-resident atomic scatter (xp/out are 51 MB each -> fit in 192 MB L2).
// a_edge folded to ea @ M (M = W_edge contracted with att_edge).
// GAT/l1/l2 biases dropped (they cancel inside the following BatchNorm).
// ---------------------------------------------------------------------------

typedef __attribute__((ext_vector_type(16))) _Float16 v16h;
typedef __attribute__((ext_vector_type(8)))  _Float16 v8h;
typedef __attribute__((ext_vector_type(8)))  float    v8f;
typedef __attribute__((ext_vector_type(4)))  float    v4f;

#define NN   50000
#define NE   800000
#define NEP  850000          // NE + NN self loops
#define EPS_BN 1e-5f
#define NEG_SLOPE 0.2f
#define CDIV(a,b) (((a)+(b)-1)/(b))

// ---- input pointer indices (recursive insertion order of setup_inputs()) ----
#define I_X    0
#define I_EI   1
#define I_EA   2
#define C1_W   3
#define C1_AS  4
#define C1_AD  5
#define C1_WE  6
#define C1_AE  7
// C1 bias (8) unused: cancels in bn1
#define C2_W   9
#define C2_AS  10
#define C2_AD  11
#define C2_WE  12
#define C2_AE  13
#define C3_W   15
#define C3_AS  16
#define C3_AD  17
#define C3_WE  18
#define C3_AE  19
#define BN1_G  21
#define BN1_B  22
#define BNL1_G 23
#define BNL1_B 24
#define BN2_G  25
#define BN2_B  26
#define BNL2_G 27
#define BNL2_B 28
#define BN3_G  29
#define BN3_B  30
#define L1_W   31
#define L2_W   33
#define LIN_W  35
#define LIN_B  36

// order-preserving float <-> uint for atomicMax-based segment max
__device__ __forceinline__ unsigned f32_ord(float f) {
  unsigned u = __float_as_uint(f);
  return (u & 0x80000000u) ? ~u : (u | 0x80000000u);
}
__device__ __forceinline__ float ord_f32(unsigned u) {
  return (u & 0x80000000u) ? __uint_as_float(u ^ 0x80000000u)
                           : __uint_as_float(~u);
}

// ---------------------------------------------------------------------------
// vectorized zero fill (n must be a multiple of 4 -- all call sites are)
__global__ void k_zero4(float* __restrict__ p, int n4) {
  int i = blockIdx.x * blockDim.x + threadIdx.x;
  if (i < n4) ((v4f*)p)[i] = (v4f){0.f, 0.f, 0.f, 0.f};
}

// dst[R2 x C2] (f16, row-major) <- src[R1 x C1] (f32), zero padded.
// 8 elements per thread (C2 % 8 == 0 at all call sites) -> b128 stores.
__global__ void k_cvt_pad_f16(const float* __restrict__ src, _Float16* __restrict__ dst,
                              int R1, int C1, int R2, int C2) {
  long long base = ((long long)blockIdx.x * blockDim.x + threadIdx.x) * 8;
  if (base >= (long long)R2 * C2) return;
  int r  = (int)(base / C2);
  int c0 = (int)(base % C2);
  v8h o;
#pragma unroll
  for (int i = 0; i < 8; ++i) {
    int c = c0 + i;
    o[i] = (r < R1 && c < C1) ? (_Float16)src[(size_t)r * C1 + c] : (_Float16)0.f;
  }
  *(v8h*)(dst + base) = o;
}

// Pack B[K x N] (f32, row-major, logical Kpad x Npad zero-padded) into WMMA
// fragment order: dst[(((nt*(Kpad/32) + kt)*32 + lane)*16 + i]
// where lane -> (half = lane>>4, c = lane&15), element i covers
// k = kt*32 + half*8 + i (i<8), kt*32 + 16 + half*8 + (i-8) (i>=8), n = nt*16 + c  (ISA 7.12.2 16-bit B layout).
// Each lane's fragment is then one contiguous 32-byte (v16h) load.
__global__ void k_pack_B(const float* __restrict__ src, _Float16* __restrict__ dst,
                         int R1, int C1, int Kpad, int Npad) {
  long long idx = (long long)blockIdx.x * blockDim.x + threadIdx.x;
  long long tot = (long long)Kpad * Npad;
  if (idx >= tot) return;
  int i    = (int)(idx & 15);
  int lane = (int)((idx >> 4) & 31);
  int t    = (int)(idx >> 9);                 // tile linear = nt*(Kpad/32) + kt
  int KT   = Kpad >> 5;
  int nt   = t / KT, kt = t - nt * KT;
  int half = lane >> 4, c = lane & 15;
  int k = kt * 32 + half * 8 + (i & 7) + ((i >> 3) << 4);
  int n = nt * 16 + c;
  dst[idx] = (k < R1 && n < C1) ? (_Float16)src[(size_t)k * C1 + n] : (_Float16)0.f;
}

// C[M,N] = A[M,K] @ B[K,N]; A row-major f16, Bp fragment-packed f16, C f32.
// M%16==0. One wave computes a 16 x (16*NT) strip. A fragment: two b128 loads,
// reused across NT accumulators. B fragments: constant-immediate-offset v16h
// loads off one per-wave base pointer. All store offsets compile-time.
template <int K, int N, int NT>
__global__ void k_gemm_wmma(const _Float16* __restrict__ A, const _Float16* __restrict__ Bp,
                            float* __restrict__ C, int M) {
  constexpr int KT = K / 32;
  int wave = (blockIdx.x * blockDim.x + threadIdx.x) >> 5;
  int lane = threadIdx.x & 31;
  constexpr int strips = (N >> 4) / NT;
  int totWaves = (M >> 4) * strips;
  if (wave >= totWaves) return;                 // wave-uniform, EXEC stays full
  int tm = wave / strips, ts = wave - tm * strips;
  int row0 = tm << 4, nt0 = ts * NT;
  int half = lane >> 4, r = lane & 15;

  v8f acc[NT];
#pragma unroll
  for (int j = 0; j < NT; ++j) acc[j] = {};

  const _Float16* arow  = A + (size_t)(row0 + r) * K + half * 8;
  const _Float16* bbase = Bp + (size_t)nt0 * KT * 512 + lane * 16;   // 512 f16 per k-tile
#pragma unroll
  for (int kt = 0; kt < KT; ++kt) {
    v8h a0 = *(const v8h*)(arow + kt * 32);
    v8h a1 = *(const v8h*)(arow + kt * 32 + 16);
    v16h af;
#pragma unroll
    for (int i = 0; i < 8; ++i) { af[i] = a0[i]; af[i + 8] = a1[i]; }
#pragma unroll
    for (int j = 0; j < NT; ++j) {
      v16h bf = *(const v16h*)(bbase + (j * KT + kt) * 512);         // const offset
      acc[j] = __builtin_amdgcn_wmma_f32_16x16x32_f16(false, af, false, bf,
                                                      (short)0, acc[j], false, false);
    }
  }
  float* cb = C + (size_t)(row0 + half * 8) * N + nt0 * 16 + r;
#pragma unroll
  for (int j = 0; j < NT; ++j)
#pragma unroll
    for (int m = 0; m < 8; ++m)                 // C: row = 8*half + m, col = lane&15
      cb[m * N + j * 16] = acc[j][m];
}

// self-loop edge_attr: mean of incoming edge_attr (PyG fill_value='mean')
__global__ void k_loop_accum(const int* __restrict__ dst, const float* __restrict__ ea,
                             float* __restrict__ cnt, float* __restrict__ lsum) {
  int idx = blockIdx.x * blockDim.x + threadIdx.x;   // (e,k) pairs
  if (idx >= NE * 12) return;
  int e = idx / 12, k = idx - e * 12;
  int d = dst[e];
  atomicAdd(&lsum[(size_t)d * 12 + k], ea[idx]);
  if (k == 0) atomicAdd(&cnt[d], 1.f);
}
__global__ void k_loop_final(float* __restrict__ lattr, const float* __restrict__ lsum,
                             const float* __restrict__ cnt) {
  int idx = blockIdx.x * blockDim.x + threadIdx.x;
  if (idx >= NN * 12) return;
  lattr[idx] = lsum[idx] / fmaxf(cnt[idx / 12], 1.f);
}

// M[k,h] = sum_c W_edge[k, h*C + c] * att_edge[h, c]   (12 x H)
__global__ void k_medge(const float* __restrict__ We, const float* __restrict__ attE,
                        float* __restrict__ Me, int H, int Cd) {
  int idx = blockIdx.x * blockDim.x + threadIdx.x;
  if (idx >= 12 * H) return;
  int k = idx / H, h = idx - k * H;
  float s = 0.f;
  for (int c = 0; c < Cd; ++c) s += We[(size_t)k * 256 + h * Cd + c] * attE[h * Cd + c];
  Me[k * H + h] = s;
}

// a_src[n,h], a_dst[n,h] from xp[n, h*C + c]
__global__ void k_node_att(const float* __restrict__ xp, const float* __restrict__ attS,
                           const float* __restrict__ attD, float* __restrict__ as_,
                           float* __restrict__ ad_, int H, int Cd) {
  int idx = blockIdx.x * blockDim.x + threadIdx.x;
  if (idx >= NN * H) return;
  int n = idx / H, h = idx - n * H;
  const float* row = xp + (size_t)n * 256 + h * Cd;
  float s = 0.f, d = 0.f;
  for (int c = 0; c < Cd; ++c) { s += row[c] * attS[h * Cd + c]; d += row[c] * attD[h * Cd + c]; }
  as_[idx] = s; ad_[idx] = d;
}

__global__ void k_init_seg(unsigned* __restrict__ amax, float* __restrict__ denom, int n) {
  int i = blockIdx.x * blockDim.x + threadIdx.x;
  if (i < n) { amax[i] = 0u; denom[i] = 0.f; }     // 0 < encode(any finite float)
}

__global__ void k_edge_alpha(const int* __restrict__ src, const int* __restrict__ dst,
                             const float* __restrict__ ea, const float* __restrict__ lattr,
                             const float* __restrict__ Me, const float* __restrict__ as_,
                             const float* __restrict__ ad_, float* __restrict__ alpha,
                             unsigned* __restrict__ amax, int H) {
  int idx = blockIdx.x * blockDim.x + threadIdx.x;
  if (idx >= NEP * H) return;
  int e = idx / H, h = idx - e * H;
  int s, d; const float* arow;
  if (e < NE) { s = src[e]; d = dst[e]; arow = ea + (size_t)e * 12; }
  else        { s = d = e - NE;         arow = lattr + (size_t)(e - NE) * 12; }
  float a = 0.f;
#pragma unroll
  for (int k = 0; k < 12; ++k) a += arow[k] * Me[k * H + h];
  float al = as_[(size_t)s * H + h] + ad_[(size_t)d * H + h] + a;
  al = (al > 0.f) ? al : NEG_SLOPE * al;
  alpha[idx] = al;
  atomicMax(&amax[(size_t)d * H + h], f32_ord(al));
}

__global__ void k_edge_exp(const int* __restrict__ dst, float* __restrict__ alpha,
                           const unsigned* __restrict__ amax, float* __restrict__ denom, int H) {
  int idx = blockIdx.x * blockDim.x + threadIdx.x;
  if (idx >= NEP * H) return;
  int e = idx / H, h = idx - e * H;
  int d = (e < NE) ? dst[e] : e - NE;
  float ex = expf(alpha[idx] - ord_f32(amax[(size_t)d * H + h]));
  alpha[idx] = ex;
  atomicAdd(&denom[(size_t)d * H + h], ex);
}

// warp per edge: out[dst, f] += xp[src, f] * ex[e, f>>logC] / (denom[dst, h]+1e-16)
__global__ void k_scatter(const int* __restrict__ src, const int* __restrict__ dst,
                          const float* __restrict__ xp, const float* __restrict__ alpha,
                          const float* __restrict__ denom, float* __restrict__ out,
                          int H, int logC) {
  int gtid = blockIdx.x * blockDim.x + threadIdx.x;
  int e = gtid >> 5, lane = gtid & 31;
  if (e >= NEP) return;
  int s, d;
  if (e < NE) { s = src[e]; d = dst[e]; } else { s = d = e - NE; }
  const float* xr = xp + (size_t)s * 256;
  float* orow = out + (size_t)d * 256;
#pragma unroll
  for (int i = 0; i < 8; ++i) {
    int f = lane + i * 32;
    int h = f >> logC;
    float w = alpha[(size_t)e * H + h] / (denom[(size_t)d * H + h] + 1e-16f);
    atomicAdd(&orow[f], xr[f] * w);
  }
}

// BatchNorm (biased var, matching jnp var ddof=0): one block per feature
__global__ void k_bn_stats(const float* __restrict__ h, float* __restrict__ mu,
                           float* __restrict__ rstd, int F) {
  __shared__ float ss[256], ss2[256];
  int f = blockIdx.x;
  float s = 0.f, s2 = 0.f;
  for (int n = threadIdx.x; n < NN; n += 256) {
    float v = h[(size_t)n * F + f];
    s += v; s2 += v * v;
  }
  ss[threadIdx.x] = s; ss2[threadIdx.x] = s2;
  __syncthreads();
  for (int o = 128; o > 0; o >>= 1) {
    if (threadIdx.x < o) { ss[threadIdx.x] += ss[threadIdx.x + o]; ss2[threadIdx.x] += ss2[threadIdx.x + o]; }
    __syncthreads();
  }
  if (threadIdx.x == 0) {
    float m = ss[0] / (float)NN;
    mu[f] = m;
    rstd[f] = rsqrtf(ss2[0] / (float)NN - m * m + EPS_BN);
  }
}

// BN apply + relu, float4 vectorized (F % 4 == 0 at all call sites)
__global__ void k_bn_relu(float* __restrict__ h, const float* __restrict__ mu,
                          const float* __restrict__ rstd, const float* __restrict__ g,
                          const float* __restrict__ b, int F) {
  long long i4 = (long long)blockIdx.x * blockDim.x + threadIdx.x;
  if (i4 * 4 >= (long long)NN * F) return;
  int f0 = (int)((i4 * 4) % F);
  v4f v = ((v4f*)h)[i4];
#pragma unroll
  for (int j = 0; j < 4; ++j) {
    int f = f0 + j;
    float t = (v[j] - mu[f]) * rstd[f] * g[f] + b[f];
    v[j] = t > 0.f ? t : 0.f;
  }
  ((v4f*)h)[i4] = v;
}

// d_out[n,0..9] = tmp[n,0..9 of 16] + bias
__global__ void k_out_bias(const float* __restrict__ tmp, const float* __restrict__ b,
                           float* __restrict__ out) {
  int i = blockIdx.x * blockDim.x + threadIdx.x;
  if (i >= NN * 10) return;
  int n = i / 10, j = i - n * 10;
  out[i] = tmp[(size_t)n * 16 + j] + b[j];
}

// ---------------------------------------------------------------------------
extern "C" void kernel_launch(void* const* d_in, const int* in_sizes, int n_in,
                              void* d_out, int out_size, void* d_ws, size_t ws_size,
                              hipStream_t stream) {
  (void)in_sizes; (void)n_in; (void)out_size; (void)ws_size;

  const float* x    = (const float*)d_in[I_X];
  const int*   ei   = (const int*)d_in[I_EI];
  const int*   srcp = ei;
  const int*   dstp = ei + NE;
  const float* ea   = (const float*)d_in[I_EA];

  // workspace bump allocator (~205 MB total)
  char* w = (char*)d_ws;
  auto alloc = [&](size_t bytes) { void* p = w; w += (bytes + 255) & ~(size_t)255; return p; };
  float*     buf0  = (float*)alloc((size_t)NN * 256 * 4);
  float*     buf1  = (float*)alloc((size_t)NN * 256 * 4);
  _Float16*  Ah    = (_Float16*)alloc((size_t)NN * 256 * 2);
  _Float16*  Bh    = (_Float16*)alloc(256 * 256 * 2);
  float*     alpha = (float*)alloc((size_t)NEP * 16 * 4);
  float*     asrc  = (float*)alloc((size_t)NN * 16 * 4);
  float*     adst  = (float*)alloc((size_t)NN * 16 * 4);
  unsigned*  amax  = (unsigned*)alloc((size_t)NN * 16 * 4);
  float*     denom = (float*)alloc((size_t)NN * 16 * 4);
  float*     cnt   = (float*)alloc((size_t)NN * 4);
  float*     lsum  = (float*)alloc((size_t)NN * 12 * 4);
  float*     lattr = (float*)alloc((size_t)NN * 12 * 4);
  float*     Me    = (float*)alloc(12 * 16 * 4);
  float*     mu    = (float*)alloc(256 * 4);
  float*     rstd  = (float*)alloc(256 * 4);
  float*     tmp16 = (float*)alloc((size_t)NN * 16 * 4);

  // ---- self-loop edge_attr mean (shared by all 3 layers) ----
  k_zero4<<<CDIV(NN / 4 + NN * 3, 256), 256, 0, stream>>>(cnt, NN / 4);          // cnt
  k_zero4<<<CDIV(NN * 12 / 4, 256), 256, 0, stream>>>(lsum, NN * 12 / 4);        // lsum
  k_loop_accum<<<CDIV(NE * 12, 256), 256, 0, stream>>>(dstp, ea, cnt, lsum);
  k_loop_final<<<CDIV(NN * 12, 256), 256, 0, stream>>>(lattr, lsum, cnt);

  auto gat = [&](const float* hin, int in_dim, int Kpad, int H, int Cd, int logC,
                 const float* W, const float* attS, const float* attD,
                 const float* We, const float* attE, float* xp, float* out) {
    k_cvt_pad_f16<<<CDIV((long long)NN * Kpad / 8, 256), 256, 0, stream>>>(hin, Ah, NN, in_dim, NN, Kpad);
    k_pack_B<<<CDIV(Kpad * 256, 256), 256, 0, stream>>>(W, Bh, in_dim, 256, Kpad, 256);
    int waves = (NN / 16) * (256 / 16 / 4);      // NT = 4
    if (Kpad == 128)
      k_gemm_wmma<128, 256, 4><<<CDIV(waves, 8), 256, 0, stream>>>(Ah, Bh, xp, NN);
    else
      k_gemm_wmma<32, 256, 4><<<CDIV(waves, 8), 256, 0, stream>>>(Ah, Bh, xp, NN);
    k_node_att<<<CDIV(NN * H, 256), 256, 0, stream>>>(xp, attS, attD, asrc, adst, H, Cd);
    k_medge<<<1, 256, 0, stream>>>(We, attE, Me, H, Cd);
    k_init_seg<<<CDIV(NN * H, 256), 256, 0, stream>>>(amax, denom, NN * H);
    k_edge_alpha<<<CDIV(NEP * H, 256), 256, 0, stream>>>(srcp, dstp, ea, lattr, Me, asrc, adst, alpha, amax, H);
    k_edge_exp<<<CDIV(NEP * H, 256), 256, 0, stream>>>(dstp, alpha, amax, denom, H);
    k_zero4<<<CDIV(NN * 64, 256), 256, 0, stream>>>(out, NN * 64);
    k_scatter<<<CDIV((long long)NEP * 32, 256), 256, 0, stream>>>(srcp, dstp, xp, alpha, denom, out, H, logC);
  };
  auto bnrelu = [&](float* h, int pg, int pb, int F) {
    k_bn_stats<<<F, 256, 0, stream>>>(h, mu, rstd, F);
    k_bn_relu<<<CDIV((long long)NN * F / 4, 256), 256, 0, stream>>>(h, mu, rstd,
        (const float*)d_in[pg], (const float*)d_in[pb], F);
  };
  auto linear = [&](const float* hin, const float* W, int Nout, int NoutPad, float* out) {
    k_cvt_pad_f16<<<CDIV((long long)NN * 256 / 8, 256), 256, 0, stream>>>(hin, Ah, NN, 256, NN, 256);
    k_pack_B<<<CDIV(256 * NoutPad, 256), 256, 0, stream>>>(W, Bh, 256, Nout, 256, NoutPad);
    int waves = NN / 16;                         // one strip covers all of N
    if (NoutPad == 16)
      k_gemm_wmma<256, 16, 1><<<CDIV(waves, 8), 256, 0, stream>>>(Ah, Bh, out, NN);
    else
      k_gemm_wmma<256, 32, 2><<<CDIV(waves, 8), 256, 0, stream>>>(Ah, Bh, out, NN);
  };

  // ---- layer 1: GAT(128 -> 16 heads x 16) ----
  gat(x, 128, 128, 16, 16, 4,
      (const float*)d_in[C1_W], (const float*)d_in[C1_AS], (const float*)d_in[C1_AD],
      (const float*)d_in[C1_WE], (const float*)d_in[C1_AE], buf1, buf0);
  bnrelu(buf0, BN1_G, BN1_B, 256);
  linear(buf0, (const float*)d_in[L1_W], 16, 16, buf1);   // bias cancels in bnl1
  bnrelu(buf1, BNL1_G, BNL1_B, 16);

  // ---- layer 2: GAT(16 -> 8 heads x 32), K padded 16 -> 32 ----
  gat(buf1, 16, 32, 8, 32, 5,
      (const float*)d_in[C2_W], (const float*)d_in[C2_AS], (const float*)d_in[C2_AD],
      (const float*)d_in[C2_WE], (const float*)d_in[C2_AE], buf0, buf1);
  bnrelu(buf1, BN2_G, BN2_B, 256);
  linear(buf1, (const float*)d_in[L2_W], 32, 32, buf0);
  bnrelu(buf0, BNL2_G, BNL2_B, 32);

  // ---- layer 3: GAT(32 -> 4 heads x 64) ----
  gat(buf0, 32, 32, 4, 64, 6,
      (const float*)d_in[C3_W], (const float*)d_in[C3_AS], (const float*)d_in[C3_AD],
      (const float*)d_in[C3_WE], (const float*)d_in[C3_AE], buf1, buf0);
  bnrelu(buf0, BN3_G, BN3_B, 256);

  // ---- classifier: [N,256] @ [256,10] (padded to 16) + bias ----
  linear(buf0, (const float*)d_in[LIN_W], 10, 16, tmp16);
  k_out_bias<<<CDIV(NN * 10, 256), 256, 0, stream>>>(tmp16, (const float*)d_in[LIN_B], (float*)d_out);
}
